// RNNModel_34411277976077
// MI455X (gfx1250) — compile-verified, matrix operand order checked by reading, not run on hardware
//
#include <hip/hip_runtime.h>

#define TSEQ 1000
#define NGATE 1536   // 2 dirs * 3H
#define NT_ALL 96    // NGATE / 16

typedef __attribute__((ext_vector_type(16))) __bf16 v16bf;
typedef __attribute__((ext_vector_type(8)))  __bf16 bf16x8;
typedef __attribute__((ext_vector_type(8)))  float  v8f;
typedef __attribute__((ext_vector_type(4))) unsigned uint32x4;
typedef __attribute__((ext_vector_type(8))) int      int32x8;
typedef __attribute__((ext_vector_type(4))) int      int32x4;

#if defined(__HIP_DEVICE_COMPILE__) && __has_builtin(__builtin_amdgcn_tensor_load_to_lds)
#define USE_TDM 1
#else
#define USE_TDM 0
#endif

union FragAB { v16bf v; bf16x8 h[2]; };

__device__ __forceinline__ float sigmoidf_(float x) {
  return 1.0f / (1.0f + __expf(-x));
}

#if USE_TDM
// Issue one TDM load of a 64x32 bf16 tile (row stride K elems) into LDS,
// hardware-padded to a 40-element LDS row stride. All descriptor words are
// wave-uniform; readfirstlane makes that provable so no waterfall is emitted.
__device__ __forceinline__ void tdm_load_tile(const __bf16* gptr,
                                              unsigned ldsoff, int K) {
  const unsigned long long ga = (unsigned long long)(uintptr_t)gptr;
  uint32x4 g0;
  g0[0] = 1u;                                           // count=1, load D#
  g0[1] = (unsigned)__builtin_amdgcn_readfirstlane((int)ldsoff);
  g0[2] = (unsigned)__builtin_amdgcn_readfirstlane((int)(unsigned)ga);
  g0[3] = ((unsigned)__builtin_amdgcn_readfirstlane((int)(unsigned)(ga >> 32))) |
          (2u << 30);                                   // type=2 ("image")
  int32x8 g1;
  g1[0] = (int)((1u << 16) |                            // data_size = 2 bytes
                (1u << 20) |                            // pad_enable
                (3u << 22) |                            // pad_interval: 16 DWORDs
                (3u << 25));                            // pad_amount: 4 DWORDs
  g1[1] = (int)(32u << 16);                             // tensor_dim0 = 32
  g1[2] = (int)(64u << 16);                             // tensor_dim1 = 64
  g1[3] = (int)(32u << 16);                             // tile_dim0 = 32
  g1[4] = 64;                                           // tile_dim1=64, tile_dim2=0
  g1[5] = __builtin_amdgcn_readfirstlane(K);            // tensor_dim0_stride
  g1[6] = 0;
  g1[7] = 0;
  const int32x4 gz4 = {0, 0, 0, 0};
  const int32x8 gz8 = {0, 0, 0, 0, 0, 0, 0, 0};
  __builtin_amdgcn_tensor_load_to_lds(g0, g1, gz4, gz4, gz8, 0);
}
#endif

// ---------------------------------------------------------------------------
// Pack weights [2, 768, K] f32 -> WMMA B-fragment-major bf16.
// Packed idx(kt, nt, chunk, lane, j) = (((kt*96 + nt)*2 + c)*32 + l)*8 + j
// maps to W element (col = nt*16 + (l&15), k = kt*32 + c*16 + (l>>4)*8 + j).
// ---------------------------------------------------------------------------
__global__ void pack_w_kernel(const float* __restrict__ w,
                              __bf16* __restrict__ dst, int K) {
  int idx  = blockIdx.x * 256 + threadIdx.x;
  int kt   = idx / 49152;          // 96*2*32*8
  int rem  = idx - kt * 49152;
  int nt   = rem >> 9;
  int rem2 = rem & 511;
  int c    = rem2 >> 8;
  int l    = (rem2 >> 3) & 31;
  int j    = rem2 & 7;
  int ncol = nt * 16 + (l & 15);
  int d    = ncol / 768;
  int n    = ncol - d * 768;
  int k    = kt * 32 + c * 16 + ((l >> 4) << 3) + j;
  dst[idx] = (__bf16)w[((size_t)(d * 768 + n)) * K + k];
}

// ---------------------------------------------------------------------------
// batch [B,T,8,40] f32 -> Xbf [T,B,320] bf16 with feat = a*8 + c (swapaxes 2,3)
// ---------------------------------------------------------------------------
__global__ void prep_x_kernel(const float* __restrict__ batch,
                              __bf16* __restrict__ xbf) {
  int idx = blockIdx.x * 256 + threadIdx.x;   // over 64000*320
  int f   = idx % 320;
  int rbt = idx / 320;
  int b   = rbt & 63;
  int t   = rbt >> 6;
  int a   = f >> 3, c = f & 7;
  xbf[idx] = (__bf16)batch[(((size_t)b * TSEQ + t) * 8 + c) * 40 + a];
}

// ---------------------------------------------------------------------------
// gi[M, 1536] = A[M, K] (bf16 row-major) @ Wp (fragment-packed bf16) + bias
// Block = 64x64 C tile, 256 threads / 8 waves, 2 tiles per wave.
// A tile staged to LDS by the Tensor Data Mover (double-buffered, TENSORcnt).
// ---------------------------------------------------------------------------
__global__ __launch_bounds__(256) void gemm_bias_kernel(
    const __bf16* __restrict__ A, int K,
    const __bf16* __restrict__ Wp,
    const float* __restrict__ bias,
    float* __restrict__ C) {
  __shared__ __bf16 As[2][64][40];   // 32 K + 8 pad per row, double-buffered
  const int tid  = threadIdx.x;
  const int lane = tid & 31;
  const int w    = tid >> 5;
  const int mt   = w & 3;
  const int ntl0 = (w >> 2) * 2;
  const int m0   = blockIdx.x * 64;
  const int ntb  = blockIdx.y * 4;

  v8f acc0 = {}; v8f acc1 = {};
  const int KT   = K >> 5;
  const int m    = mt * 16 + (lane & 15);
  const int kb   = (lane >> 4) << 3;
#if USE_TDM
  const unsigned ldsoff0 = (unsigned)(uintptr_t)(&As[0][0][0]);
  const unsigned ldsoff1 = (unsigned)(uintptr_t)(&As[1][0][0]);
  const __bf16* arow = A + (size_t)m0 * (unsigned)K;
#else
  const int lrow = tid >> 2;
  const int lcol = (tid & 3) << 3;
#endif

  for (int kt = 0; kt < KT; ++kt) {
#if USE_TDM
    if (w == 0) {
      if (kt == 0) tdm_load_tile(arow, ldsoff0, K);
      if (kt + 1 < KT) {
        tdm_load_tile(arow + (kt + 1) * 32, ((kt + 1) & 1) ? ldsoff1 : ldsoff0,
                      K);
        __builtin_amdgcn_s_wait_tensorcnt(1);  // tile kt done, kt+1 in flight
      } else {
        __builtin_amdgcn_s_wait_tensorcnt(0);  // last tile done
      }
    }
    __syncthreads();   // tile kt visible to all waves
#else
    bf16x8 av = *reinterpret_cast<const bf16x8*>(
        A + (size_t)(m0 + lrow) * K + kt * 32 + lcol);
    *reinterpret_cast<bf16x8*>(&As[kt & 1][lrow][lcol]) = av;
    __syncthreads();
#endif

    FragAB af;
    af.h[0] = *reinterpret_cast<const bf16x8*>(&As[kt & 1][m][kb]);
    af.h[1] = *reinterpret_cast<const bf16x8*>(&As[kt & 1][m][kb + 16]);

    const __bf16* wpk =
        Wp + ((size_t)kt * NT_ALL + (ntb + ntl0)) * 512 + lane * 8;
    FragAB bf0, bf1;
    bf0.h[0] = *reinterpret_cast<const bf16x8*>(wpk);
    bf0.h[1] = *reinterpret_cast<const bf16x8*>(wpk + 256);
    bf1.h[0] = *reinterpret_cast<const bf16x8*>(wpk + 512);
    bf1.h[1] = *reinterpret_cast<const bf16x8*>(wpk + 768);

    acc0 = __builtin_amdgcn_wmma_f32_16x16x32_bf16(
        false, af.v, false, bf0.v, (short)0, acc0, false, false);
    acc1 = __builtin_amdgcn_wmma_f32_16x16x32_bf16(
        false, af.v, false, bf1.v, (short)0, acc1, false, false);
    __syncthreads();   // all reads of buffer (kt&1) done before it is refilled
  }

  const int g  = lane >> 4;
  const int nc = lane & 15;
  const float b0v = bias[(ntb + ntl0) * 16 + nc];
  const float b1v = bias[(ntb + ntl0 + 1) * 16 + nc];
#pragma unroll
  for (int r = 0; r < 8; ++r) {
    const size_t row = (size_t)(m0 + mt * 16 + r + 8 * g);
    C[row * NGATE + (ntb + ntl0) * 16 + nc]     = acc0[r] + b0v;
    C[row * NGATE + (ntb + ntl0 + 1) * 16 + nc] = acc1[r] + b1v;
  }
}

// ---------------------------------------------------------------------------
// Persistent bidirectional GRU recurrence. grid = {2} (dir), 1024 threads.
// h state in LDS (bf16); gh = h @ Whh^T via WMMA each step; gate math f32.
// ---------------------------------------------------------------------------
__global__ __launch_bounds__(1024) void gru_recur_kernel(
    const float* __restrict__ gi,     // [T*64*1536] (both dirs stacked)
    const __bf16* __restrict__ WhhP,  // packed, NT=96 over both dirs
    const float* __restrict__ bhh,    // [2*768]
    __bf16* __restrict__ outH) {      // [T*64*512]
  __shared__ __bf16 hB[64][264];      // h state, padded (33 KB)
  __shared__ __bf16 gh[64][768];      // gate pre-activations (96 KB)

  const int d    = blockIdx.x;
  const int tid  = threadIdx.x;
  const int lane = tid & 31;
  const int wv   = tid >> 5;
  const int mt   = wv >> 3;           // 0..3 : M tile
  const int nb   = wv & 7;            // 0..7 : N tile group (6 tiles each)

  for (int i = tid; i < 64 * 264; i += 1024) (&hB[0][0])[i] = (__bf16)0.0f;
  __syncthreads();

  const int m  = mt * 16 + (lane & 15);
  const int kb = (lane >> 4) << 3;
  const int gb = tid >> 4;            // batch row for gate phase
  const int jb = (tid & 15) << 4;     // column base for gate phase
  const float* bh = bhh + d * 768;

  for (int it = 0; it < TSEQ; ++it) {
    const int t = d ? (TSEQ - 1 - it) : it;
    v8f acc[6] = {};
#pragma unroll 2
    for (int kt = 0; kt < 8; ++kt) {
      FragAB af;
      af.h[0] = *reinterpret_cast<const bf16x8*>(&hB[m][kt * 32 + kb]);
      af.h[1] = *reinterpret_cast<const bf16x8*>(&hB[m][kt * 32 + kb + 16]);
      __builtin_prefetch(
          WhhP + ((size_t)(((kt + 1) & 7) * 96 + d * 48 + nb)) * 512 + lane * 8,
          0, 1);
#pragma unroll
      for (int i = 0; i < 6; ++i) {
        const int nt = d * 48 + nb + 8 * i;
        const __bf16* wp = WhhP + ((size_t)(kt * 96 + nt)) * 512 + lane * 8;
        FragAB bfr;
        bfr.h[0] = *reinterpret_cast<const bf16x8*>(wp);
        bfr.h[1] = *reinterpret_cast<const bf16x8*>(wp + 256);
        acc[i] = __builtin_amdgcn_wmma_f32_16x16x32_bf16(
            false, af.v, false, bfr.v, (short)0, acc[i], false, false);
      }
    }
#pragma unroll
    for (int i = 0; i < 6; ++i) {
      const int n = (nb + 8 * i) * 16 + (lane & 15);
#pragma unroll
      for (int r = 0; r < 8; ++r)
        gh[mt * 16 + r + 8 * (lane >> 4)][n] = (__bf16)acc[i][r];
    }
    __syncthreads();

    const float* girow = gi + ((size_t)t * 64 + gb) * NGATE + d * 768;
    __bf16* orow = outH + ((size_t)t * 64 + gb) * 512 + d * 256;
#pragma unroll 4
    for (int jj = 0; jj < 16; ++jj) {
      const int j = jb + jj;
      const float gr = (float)gh[gb][j]       + bh[j];
      const float gz = (float)gh[gb][256 + j] + bh[256 + j];
      const float gn = (float)gh[gb][512 + j] + bh[512 + j];
      const float r  = sigmoidf_(girow[j] + gr);
      const float z  = sigmoidf_(girow[256 + j] + gz);
      const float n  = tanhf(girow[512 + j] + r * gn);
      const float ho = (float)hB[gb][j];
      const float hn = (1.0f - z) * n + z * ho;
      hB[gb][j] = (__bf16)hn;
      orow[j]   = (__bf16)hn;
    }
    __syncthreads();
  }
}

// ---------------------------------------------------------------------------
// FC (512 -> 61) + variable-length gather into packed output rows.
// ---------------------------------------------------------------------------
__global__ __launch_bounds__(64) void fc_gather_kernel(
    const __bf16* __restrict__ h2, const float* __restrict__ fcw,
    const float* __restrict__ fcb, const int* __restrict__ lengths,
    float* __restrict__ out) {
  const int bi = blockIdx.x;
  const int b  = bi / TSEQ;
  const int t  = bi - b * TSEQ;
  __shared__ int offs;
  if (threadIdx.x == 0) {
    int s = 0;
    for (int i = 0; i < b; ++i) s += lengths[i];
    offs = s;
  }
  __syncthreads();
  if (t >= lengths[b]) return;
  const int o = threadIdx.x;
  if (o >= 61) return;
  const __bf16* hrow = h2 + ((size_t)t * 64 + b) * 512;
  const float* wrow  = fcw + (size_t)o * 512;
  float acc = fcb[o];
  for (int k = 0; k < 512; k += 4) {
    acc += (float)hrow[k]     * wrow[k]     + (float)hrow[k + 1] * wrow[k + 1] +
           (float)hrow[k + 2] * wrow[k + 2] + (float)hrow[k + 3] * wrow[k + 3];
  }
  out[(size_t)(offs + t) * 61 + o] = acc;
}

// ---------------------------------------------------------------------------
extern "C" void kernel_launch(void* const* d_in, const int* in_sizes, int n_in,
                              void* d_out, int out_size, void* d_ws,
                              size_t ws_size, hipStream_t stream) {
  (void)in_sizes; (void)n_in; (void)out_size; (void)ws_size;
  const float* batch   = (const float*)d_in[0];
  const int*   lengths = (const int*)d_in[1];
  const float* w_ih_l0 = (const float*)d_in[2];
  const float* w_hh_l0 = (const float*)d_in[3];
  const float* b_ih_l0 = (const float*)d_in[4];
  const float* b_hh_l0 = (const float*)d_in[5];
  const float* w_ih_l1 = (const float*)d_in[6];
  const float* w_hh_l1 = (const float*)d_in[7];
  const float* b_ih_l1 = (const float*)d_in[8];
  const float* b_hh_l1 = (const float*)d_in[9];
  const float* fc_w    = (const float*)d_in[10];
  const float* fc_b    = (const float*)d_in[11];
  float* out = (float*)d_out;

  char* ws = (char*)d_ws;
  size_t off = 0;
  auto take = [&](size_t bytes) -> char* {
    char* p = ws + off;
    off = (off + bytes + 255) & ~(size_t)255;
    return p;
  };
  __bf16* Xbf   = (__bf16*)take((size_t)64000 * 320 * 2);
  __bf16* h1    = (__bf16*)take((size_t)64000 * 512 * 2);
  __bf16* h2    = (__bf16*)take((size_t)64000 * 512 * 2);
  float*  gi    = (float*)take((size_t)64000 * 1536 * 4);  // reused by both layers
  __bf16* WihP0 = (__bf16*)take((size_t)10 * 96 * 512 * 2);
  __bf16* WhhP0 = (__bf16*)take((size_t)8  * 96 * 512 * 2);
  __bf16* WihP1 = (__bf16*)take((size_t)16 * 96 * 512 * 2);
  __bf16* WhhP1 = (__bf16*)take((size_t)8  * 96 * 512 * 2);

  // Weight packing (tiny, one-shot per call)
  pack_w_kernel<<<491520 / 256, 256, 0, stream>>>(w_ih_l0, WihP0, 320);
  pack_w_kernel<<<393216 / 256, 256, 0, stream>>>(w_hh_l0, WhhP0, 256);
  pack_w_kernel<<<786432 / 256, 256, 0, stream>>>(w_ih_l1, WihP1, 512);
  pack_w_kernel<<<393216 / 256, 256, 0, stream>>>(w_hh_l1, WhhP1, 256);

  // Input reshape + bf16
  prep_x_kernel<<<80000, 256, 0, stream>>>(batch, Xbf);

  // Layer 0: input projection (all T at once), then persistent recurrence
  gemm_bias_kernel<<<dim3(1000, 24), 256, 0, stream>>>(Xbf, 320, WihP0,
                                                       b_ih_l0, gi);
  gru_recur_kernel<<<2, 1024, 0, stream>>>(gi, WhhP0, b_hh_l0, h1);

  // Layer 1
  gemm_bias_kernel<<<dim3(1000, 24), 256, 0, stream>>>(h1, 512, WihP1,
                                                       b_ih_l1, gi);
  gru_recur_kernel<<<2, 1024, 0, stream>>>(gi, WhhP1, b_hh_l1, h2);

  // FC + length gather
  fc_gather_kernel<<<64000, 64, 0, stream>>>(h2, fc_w, fc_b, lengths, out);
}
